// GNN_Origin_57509612093942
// MI455X (gfx1250) — matrix operand order repaired and checked by reading, not compiled, not run on hardware
//
#include <hip/hip_runtime.h>
#include <hip/hip_bf16.h>

// ---------------------------------------------------------------------------
// SAGPool GNN forward, CDNA5 (gfx1250).
// Heavy GEMMs: one wave = one 16x128 row-stripe, A-fragment reused across
// 8x v_wmma_f32_16x16x32_f16; B operand (XW / W) is kept pre-converted f16 and
// pre-transposed so each B fragment is two 16B vector loads.  Adjacency
// (the roofline-dominant 256MB array) is streamed in full three times (deg, agg, score).
// ---------------------------------------------------------------------------

typedef __attribute__((ext_vector_type(16))) _Float16 v16h;
typedef __attribute__((ext_vector_type(8)))  _Float16 v8h;
typedef __attribute__((ext_vector_type(8)))  float    v8f;
typedef __attribute__((ext_vector_type(4)))  float    v4f;

#define BB   256
#define NN   512
#define HH   128
#define KK1  103
#define KK2  21
#define KK3  5
#define NPAD1 512
#define NPAD2 128
#define NPAD3 32
#define NEG_INF (-3.4e38f)

__device__ __forceinline__ v8f wmma_f16(v16h a, v16h b, v8f c) {
  // v_wmma_f32_16x16x32_f16: D = A(16x32 f16) x B(32x16 f16) + C(16x16 f32)
  return __builtin_amdgcn_wmma_f32_16x16x32_f16(false, a, false, b,
                                                (short)0, c, false, false);
}

__device__ __forceinline__ float waveReduceSum(float v) {
#pragma unroll
  for (int off = 16; off; off >>= 1) v += __shfl_xor(v, off, 32);
  return v;
}

// ---------------- WT[c][k] = (f16) W[k][c]  (128x128, one-shot) -------------
__global__ void wt_kernel(const float* __restrict__ W, _Float16* __restrict__ WT) {
  const int gid = blockIdx.x * blockDim.x + threadIdx.x;
  if (gid >= HH * HH) return;
  const int c = gid >> 7, k = gid & (HH - 1);
  WT[gid] = (_Float16)W[(size_t)k * HH + c];
}

// ---------------- degree: d[b,i] = rsqrt(1 + sum_j adj[b,i,j]) --------------
__global__ void deg_kernel(const float* __restrict__ adj, float* __restrict__ dvec,
                           int rows, int n) {
  const int wid  = blockIdx.x * (blockDim.x >> 5) + (threadIdx.x >> 5);
  const int lane = threadIdx.x & 31;
  if (wid >= rows) return;
  const float* row = adj + (size_t)wid * n;
  float sum = 0.f;
  for (int j = lane; j < n; j += 32) sum += row[j];
  sum = waveReduceSum(sum);
  if (lane == 0) dvec[wid] = rsqrtf(sum + 1.0f);
}

// ---------------- xwT[c][b][lr] = (f16)(X @ W)[b*n+lr][c] -------------------
// X:[Mtot,128] f32 row-major; WT: f16 [128(c)][128(k)]; output f16 transposed,
// per-graph rows padded to npad.  One wave = 16x128 stripe, 8 wmma per K-step.
__global__ void xw_gemm_kernel(const float* __restrict__ X, const _Float16* __restrict__ WT,
                               _Float16* __restrict__ xwT, int Mtot, int n, int npad) {
  const int lane = threadIdx.x & 31;
  const int wid  = blockIdx.x * (blockDim.x >> 5) + (threadIdx.x >> 5);
  if (wid >= (Mtot >> 4)) return;
  const int mbase = wid << 4;
  const int g = lane >> 4, l = lane & 15;
  const size_t cstride = (size_t)BB * npad;

  const float* xrow = X + (size_t)(mbase + l) * HH;
  v8f acc[8] = {};
  for (int kk = 0; kk < HH; kk += 32) {
    float af[16] __attribute__((aligned(16)));
    // A lane layout identity: a[p] holds K = kk + (p>>3)*16 + g*8 + (p&7)
    *(v4f*)(af + 0)  = *(const v4f*)(xrow + kk + (g << 3));
    *(v4f*)(af + 4)  = *(const v4f*)(xrow + kk + (g << 3) + 4);
    *(v4f*)(af + 8)  = *(const v4f*)(xrow + kk + 16 + (g << 3));
    *(v4f*)(af + 12) = *(const v4f*)(xrow + kk + 16 + (g << 3) + 4);
    v16h a;
#pragma unroll
    for (int p = 0; p < 16; ++p) a[p] = (_Float16)af[p];
#pragma unroll
    for (int nt = 0; nt < 8; ++nt) {
      const int c = (nt << 4) + l;
      // B fragment: bm[i] <-> K = kk + (i>>3)*16 + g*8 + (i&7) : two 8xf16 loads
      const v8h b0 = *(const v8h*)(WT + (size_t)c * HH + kk + (g << 3));
      const v8h b1 = *(const v8h*)(WT + (size_t)c * HH + kk + 16 + (g << 3));
      const v16h bm = __builtin_shufflevector(b0, b1, 0, 1, 2, 3, 4, 5, 6, 7, 8, 9, 10, 11, 12, 13, 14, 15);
      acc[nt] = wmma_f16(a, bm, acc[nt]);
    }
  }
  if (n == npad) {              // stripes never cross graphs: vector store
#pragma unroll
    for (int nt = 0; nt < 8; ++nt) {
      const int c = (nt << 4) + l;
      v8h o;
#pragma unroll
      for (int r = 0; r < 8; ++r) o[r] = (_Float16)acc[nt][r];
      *(v8h*)(xwT + (size_t)c * cstride + mbase + (g << 3)) = o;
    }
  } else {
    int rb[8], ro[8];
#pragma unroll
    for (int r = 0; r < 8; ++r) {
      const int row = mbase + (g << 3) + r;
      rb[r] = row / n; ro[r] = row % n;
    }
#pragma unroll
    for (int nt = 0; nt < 8; ++nt) {
      const int c = (nt << 4) + l;
      _Float16* col = xwT + (size_t)c * cstride;
#pragma unroll
      for (int r = 0; r < 8; ++r)
        col[(size_t)rb[r] * npad + ro[r]] = (_Float16)acc[nt][r];
    }
  }
}

// ---- Hout[b,i,:] = relu( sum_j d_i d_j (adj[b,i,j]+delta_ij) XW[b,j,:] + b ) ----
// One wave computes a 16x128 row-stripe of Hout; the adjacency tile is built
// once per K-step (vector loads when n%4==0) and reused for 8 wmma; the B
// fragment is two 16B f16 vector loads from the transposed padded XW.
__global__ void agg_gemm_kernel(const float* __restrict__ adj, const float* __restrict__ dvec,
                                const _Float16* __restrict__ xwT, const float* __restrict__ bias,
                                float* __restrict__ out, int Bn, int n, int npad) {
  const int lane = threadIdx.x & 31;
  const int wid  = blockIdx.x * (blockDim.x >> 5) + (threadIdx.x >> 5);
  const int MT = (n + 15) >> 4;
  if (wid >= Bn * MT) return;
  const int mt = wid % MT;
  const int b  = wid / MT;
  const int mbase = mt << 4;
  const int g = lane >> 4, l = lane & 15;
  const size_t cstride = (size_t)BB * npad;

  const float* A  = adj  + (size_t)b * n * n;
  const float* dv = dvec + (size_t)b * n;
  const _Float16* Xb = xwT + (size_t)b * npad;

  const int i = mbase + l;
  const int ic = (i < n) ? i : (n - 1);
  const float di = (i < n) ? dv[i] : 0.f;
  const float* arow = A + (size_t)ic * n;
  const bool nvec = ((n & 3) == 0);

  v8f acc[8] = {};
  const int ksteps = (n + 31) >> 5;
  for (int ks = 0; ks < ksteps; ++ks) {
    const int kk = ks << 5;
    if (kk + 32 < n) __builtin_prefetch(arow + kk + 32, 0, 0);
    float af[16] __attribute__((aligned(16)));
    float df[16] __attribute__((aligned(16)));
    if (nvec && (kk + 32 <= n)) {
      *(v4f*)(af + 0)  = *(const v4f*)(arow + kk + (g << 3));
      *(v4f*)(af + 4)  = *(const v4f*)(arow + kk + (g << 3) + 4);
      *(v4f*)(af + 8)  = *(const v4f*)(arow + kk + 16 + (g << 3));
      *(v4f*)(af + 12) = *(const v4f*)(arow + kk + 16 + (g << 3) + 4);
      *(v4f*)(df + 0)  = *(const v4f*)(dv + kk + (g << 3));
      *(v4f*)(df + 4)  = *(const v4f*)(dv + kk + (g << 3) + 4);
      *(v4f*)(df + 8)  = *(const v4f*)(dv + kk + 16 + (g << 3));
      *(v4f*)(df + 12) = *(const v4f*)(dv + kk + 16 + (g << 3) + 4);
    } else {
#pragma unroll
      for (int p = 0; p < 16; ++p) {
        const int j = kk + ((p >> 3) << 4) + (g << 3) + (p & 7);
        af[p] = (i < n && j < n) ? arow[j] : 0.f;
        df[p] = (j < n) ? dv[j] : 0.f;
      }
    }
    v16h a;
#pragma unroll
    for (int p = 0; p < 16; ++p) {
      const int j = kk + ((p >> 3) << 4) + (g << 3) + (p & 7);
      const float aval = af[p] + ((j == i) ? 1.f : 0.f);   // self-loop; di==0 kills pad rows
      a[p] = (_Float16)(di * df[p] * aval);
    }
#pragma unroll
    for (int nt = 0; nt < 8; ++nt) {
      const int c = (nt << 4) + l;
      const v8h b0 = *(const v8h*)(Xb + (size_t)c * cstride + kk + (g << 3));
      const v8h b1 = *(const v8h*)(Xb + (size_t)c * cstride + kk + 16 + (g << 3));
      const v16h bm = __builtin_shufflevector(b0, b1, 0, 1, 2, 3, 4, 5, 6, 7, 8, 9, 10, 11, 12, 13, 14, 15);
      acc[nt] = wmma_f16(a, bm, acc[nt]);
    }
  }
#pragma unroll
  for (int nt = 0; nt < 8; ++nt) {
    const int nbase = nt << 4;
    const float bsv = bias[nbase + l];
#pragma unroll
    for (int r = 0; r < 8; ++r) {
      const int row = mbase + r + (g << 3);
      if (row < n) {
        const float vo = acc[nt][r] + bsv;
        out[((size_t)b * n + row) * HH + nbase + l] = vo > 0.f ? vo : 0.f;
      }
    }
  }
}

// ---------------- t = h . pr ,  u = h . po  (per row) -----------------------
__global__ void score_tu_kernel(const float* __restrict__ h, const float* __restrict__ pr,
                                const float* __restrict__ po, float* __restrict__ t,
                                float* __restrict__ u, int rows) {
  const int wid  = blockIdx.x * (blockDim.x >> 5) + (threadIdx.x >> 5);
  const int lane = threadIdx.x & 31;
  if (wid >= rows) return;
  const float* hr = h + (size_t)wid * HH;
  float st = 0.f, su = 0.f;
  for (int j = lane; j < HH; j += 32) {
    const float hv = hr[j];
    st += hv * pr[j];
    su += hv * po[j];
  }
  st = waveReduceSum(st);
  su = waveReduceSum(su);
  if (lane == 0) { t[wid] = st; u[wid] = su; }
}

// ---------------- s[b,i] = sum_j adj[b,i,j] t[b,j] + u[b,i] + pb ------------
__global__ void score_final_kernel(const float* __restrict__ adj, const float* __restrict__ t,
                                   const float* __restrict__ u, const float* __restrict__ pb,
                                   float* __restrict__ s, int rows, int n) {
  const int wid  = blockIdx.x * (blockDim.x >> 5) + (threadIdx.x >> 5);
  const int lane = threadIdx.x & 31;
  if (wid >= rows) return;
  const int b = wid / n;
  const float* arow = adj + (size_t)wid * n;
  const float* tb = t + (size_t)b * n;
  float sum = 0.f;
  for (int j = lane; j < n; j += 32) sum += arow[j] * tb[j];
  sum = waveReduceSum(sum);
  if (lane == 0) s[wid] = sum + u[wid] + pb[0];
}

// ---------------- top-k per graph (desc, ties -> lowest index) --------------
__global__ void topk_kernel(const float* __restrict__ s, float* __restrict__ vals,
                            int* __restrict__ idx, int n, int k) {
  __shared__ float sv[NN];
  __shared__ float rv[256];
  __shared__ int   ri[256];
  const int b = blockIdx.x, tid = threadIdx.x;
  for (int j = tid; j < n; j += blockDim.x) sv[j] = s[(size_t)b * n + j];
  __syncthreads();
  for (int kk = 0; kk < k; ++kk) {
    float best = NEG_INF; int bi = n;
    for (int j = tid; j < n; j += blockDim.x) {
      const float v = sv[j];
      if (v > best || (v == best && j < bi)) { best = v; bi = j; }
    }
    rv[tid] = best; ri[tid] = bi;
    __syncthreads();
    for (int off = blockDim.x >> 1; off > 0; off >>= 1) {
      if (tid < off) {
        const float ov = rv[tid + off]; const int oi = ri[tid + off];
        if (ov > rv[tid] || (ov == rv[tid] && oi < ri[tid])) { rv[tid] = ov; ri[tid] = oi; }
      }
      __syncthreads();
    }
    if (tid == 0) {
      vals[(size_t)b * k + kk] = rv[0];
      idx [(size_t)b * k + kk] = ri[0];
      sv[ri[0]] = NEG_INF;
    }
    __syncthreads();
  }
}

// ---------------- hk[b,kk,:] = h[b,idx,:] * tanh(vals) ----------------------
__global__ void gather_h_kernel(const float* __restrict__ h, const int* __restrict__ idx,
                                const float* __restrict__ vals, float* __restrict__ hk,
                                int n, int k) {
  const int gid = blockIdx.x * blockDim.x + threadIdx.x;
  const int total = BB * k * HH;
  if (gid >= total) return;
  const int hc = gid & (HH - 1);
  const int tmp = gid >> 7;
  const int kk = tmp % k;
  const int b  = tmp / k;
  const int node = idx[(size_t)b * k + kk];
  const float th = tanhf(vals[(size_t)b * k + kk]);
  hk[gid] = h[((size_t)b * n + node) * HH + hc] * th;
}

// ---------------- ak[b,i,j] = adj[b, idx[i], idx[j]] ------------------------
__global__ void gather_a_kernel(const float* __restrict__ adj, const int* __restrict__ idx,
                                float* __restrict__ ak, int n, int k) {
  const int gid = blockIdx.x * blockDim.x + threadIdx.x;
  const int total = BB * k * k;
  if (gid >= total) return;
  const int j = gid % k;
  int tmp = gid / k;
  const int i = tmp % k;
  const int b = tmp / k;
  const int ni = idx[(size_t)b * k + i];
  const int nj = idx[(size_t)b * k + j];
  ak[gid] = adj[((size_t)b * n + ni) * n + nj];
}

// ---------------- r[b,:128] += max_k hk ; r[b,128:] += mean_k hk ------------
__global__ void readout_kernel(const float* __restrict__ hk, float* __restrict__ r, int k) {
  const int gid = blockIdx.x * blockDim.x + threadIdx.x;
  if (gid >= BB * HH) return;
  const int hc = gid & (HH - 1);
  const int b  = gid >> 7;
  const float* base = hk + ((size_t)b * k) * HH + hc;
  float mx = NEG_INF, sm = 0.f;
  for (int kk = 0; kk < k; ++kk) {
    const float v = base[(size_t)kk * HH];
    mx = v > mx ? v : mx;
    sm += v;
  }
  r[(size_t)b * 2 * HH + hc]      += mx;
  r[(size_t)b * 2 * HH + HH + hc] += sm / (float)k;
}

__global__ void zero_kernel(float* __restrict__ p, int nElem) {
  const int gid = blockIdx.x * blockDim.x + threadIdx.x;
  if (gid < nElem) p[gid] = 0.f;
}

// ---------------- MLP head + log_softmax, one block per graph ---------------
__global__ void mlp_kernel(const float* __restrict__ r,
                           const float* __restrict__ l1w, const float* __restrict__ l1b,
                           const float* __restrict__ l2w, const float* __restrict__ l2b,
                           const float* __restrict__ l3w, const float* __restrict__ l3b,
                           float* __restrict__ out) {
  __shared__ float zin[2 * HH];
  __shared__ float z1[HH];
  __shared__ float z2[HH / 2];
  __shared__ float z3[16];
  const int b = blockIdx.x, tid = threadIdx.x;   // blockDim = 256
  zin[tid] = r[(size_t)b * 2 * HH + tid];
  __syncthreads();
  if (tid < HH) {
    float acc = l1b[tid];
    for (int i = 0; i < 2 * HH; ++i) acc += zin[i] * l1w[(size_t)i * HH + tid];
    z1[tid] = acc > 0.f ? acc : 0.f;
  }
  __syncthreads();
  if (tid < HH / 2) {
    float acc = l2b[tid];
    for (int i = 0; i < HH; ++i) acc += z1[i] * l2w[(size_t)i * (HH / 2) + tid];
    z2[tid] = acc > 0.f ? acc : 0.f;
  }
  __syncthreads();
  if (tid < 10) {
    float acc = l3b[tid];
    for (int i = 0; i < HH / 2; ++i) acc += z2[i] * l3w[(size_t)i * 10 + tid];
    z3[tid] = acc;
  }
  __syncthreads();
  if (tid == 0) {
    float m = z3[0];
    for (int i = 1; i < 10; ++i) m = z3[i] > m ? z3[i] : m;
    float se = 0.f;
    for (int i = 0; i < 10; ++i) se += expf(z3[i] - m);
    const float lse = m + logf(se);
    for (int i = 0; i < 10; ++i) out[(size_t)b * 10 + i] = z3[i] - lse;
  }
}

// ---------------------------------------------------------------------------
extern "C" void kernel_launch(void* const* d_in, const int* in_sizes, int n_in,
                              void* d_out, int out_size, void* d_ws, size_t ws_size,
                              hipStream_t stream) {
  (void)in_sizes; (void)n_in; (void)out_size; (void)ws_size;
  const float* x   = (const float*)d_in[0];
  const float* adj = (const float*)d_in[1];
  const float* W1  = (const float*)d_in[2];
  const float* b1  = (const float*)d_in[3];
  const float* W2  = (const float*)d_in[4];
  const float* b2  = (const float*)d_in[5];
  const float* W3  = (const float*)d_in[6];
  const float* b3  = (const float*)d_in[7];
  const float* pr1 = (const float*)d_in[8];
  const float* po1 = (const float*)d_in[9];
  const float* pb1 = (const float*)d_in[10];
  const float* pr2 = (const float*)d_in[11];
  const float* po2 = (const float*)d_in[12];
  const float* pb2 = (const float*)d_in[13];
  const float* pr3 = (const float*)d_in[14];
  const float* po3 = (const float*)d_in[15];
  const float* pb3 = (const float*)d_in[16];
  const float* l1w = (const float*)d_in[17];
  const float* l1b = (const float*)d_in[18];
  const float* l2w = (const float*)d_in[19];
  const float* l2b = (const float*)d_in[20];
  const float* l3w = (const float*)d_in[21];
  const float* l3b = (const float*)d_in[22];
  float* out = (float*)d_out;

  // --- workspace carving (floats, 32B-aligned chunks) ---
  float* ws = (float*)d_ws;
  size_t off = 0;
  auto alloc = [&](size_t nfl) { nfl = (nfl + 7) & ~(size_t)7;
                                 float* p = ws + off; off += nfl; return p; };
  _Float16* xwT = (_Float16*)alloc(((size_t)HH * BB * NPAD1) / 2 + 64); // f16 B operand
  _Float16* WT  = (_Float16*)alloc((size_t)HH * HH / 2);                // f16 W^T
  float* hbuf = alloc((size_t)BB * NN * HH);    // post-GCN features (reused)
  float* dvec = alloc((size_t)BB * NN);
  float* t    = alloc((size_t)BB * NN);
  float* u    = alloc((size_t)BB * NN);
  float* s    = alloc((size_t)BB * NN);
  float* vals = alloc((size_t)BB * KK1);
  int*   idx  = (int*)alloc((size_t)BB * KK1);
  float* hk1  = alloc((size_t)BB * KK1 * HH);
  float* ak1  = alloc((size_t)BB * KK1 * KK1);
  float* hk2  = alloc((size_t)BB * KK2 * HH);
  float* ak2  = alloc((size_t)BB * KK2 * KK2);
  float* hk3  = alloc((size_t)BB * KK3 * HH);
  float* ak3  = alloc((size_t)BB * KK3 * KK3);
  float* rr   = alloc((size_t)BB * 2 * HH);

  const int TPB = 256;
  auto rowBlocks  = [](int nrows) { return (nrows + 7) / 8; };      // wave-per-row
  auto thrBlocks  = [](int nthr)  { return (nthr + 255) / 256; };   // thread-per-elem
  auto waveBlocks = [](int nwaves){ return (nwaves + 7) / 8; };     // wave-per-stripe

  zero_kernel<<<thrBlocks(BB * 2 * HH), TPB, 0, stream>>>(rr, BB * 2 * HH);

  // ================= Stage 1: n=512, keep K1=103 =================
  {
    const int n = NN, k = KK1, npad = NPAD1;
    wt_kernel<<<thrBlocks(HH * HH), TPB, 0, stream>>>(W1, WT);
    deg_kernel<<<rowBlocks(BB * n), TPB, 0, stream>>>(adj, dvec, BB * n, n);
    xw_gemm_kernel<<<waveBlocks(BB * n / 16), TPB, 0, stream>>>(x, WT, xwT, BB * n, n, npad);
    const int MT = (n + 15) / 16;
    agg_gemm_kernel<<<waveBlocks(BB * MT), TPB, 0, stream>>>(adj, dvec, xwT, b1, hbuf, BB, n, npad);
    score_tu_kernel<<<rowBlocks(BB * n), TPB, 0, stream>>>(hbuf, pr1, po1, t, u, BB * n);
    score_final_kernel<<<rowBlocks(BB * n), TPB, 0, stream>>>(adj, t, u, pb1, s, BB * n, n);
    topk_kernel<<<BB, TPB, 0, stream>>>(s, vals, idx, n, k);
    gather_h_kernel<<<thrBlocks(BB * k * HH), TPB, 0, stream>>>(hbuf, idx, vals, hk1, n, k);
    gather_a_kernel<<<thrBlocks(BB * k * k), TPB, 0, stream>>>(adj, idx, ak1, n, k);
    readout_kernel<<<thrBlocks(BB * HH), TPB, 0, stream>>>(hk1, rr, k);
  }

  // ================= Stage 2: n=103, keep K2=21 =================
  {
    const int n = KK1, k = KK2, npad = NPAD2;
    wt_kernel<<<thrBlocks(HH * HH), TPB, 0, stream>>>(W2, WT);
    deg_kernel<<<rowBlocks(BB * n), TPB, 0, stream>>>(ak1, dvec, BB * n, n);
    xw_gemm_kernel<<<waveBlocks(BB * n / 16), TPB, 0, stream>>>(hk1, WT, xwT, BB * n, n, npad);
    const int MT = (n + 15) / 16;
    agg_gemm_kernel<<<waveBlocks(BB * MT), TPB, 0, stream>>>(ak1, dvec, xwT, b2, hbuf, BB, n, npad);
    score_tu_kernel<<<rowBlocks(BB * n), TPB, 0, stream>>>(hbuf, pr2, po2, t, u, BB * n);
    score_final_kernel<<<rowBlocks(BB * n), TPB, 0, stream>>>(ak1, t, u, pb2, s, BB * n, n);
    topk_kernel<<<BB, TPB, 0, stream>>>(s, vals, idx, n, k);
    gather_h_kernel<<<thrBlocks(BB * k * HH), TPB, 0, stream>>>(hbuf, idx, vals, hk2, n, k);
    gather_a_kernel<<<thrBlocks(BB * k * k), TPB, 0, stream>>>(ak1, idx, ak2, n, k);
    readout_kernel<<<thrBlocks(BB * HH), TPB, 0, stream>>>(hk2, rr, k);
  }

  // ================= Stage 3: n=21, keep K3=5 =================
  {
    const int n = KK2, k = KK3, npad = NPAD3;
    wt_kernel<<<thrBlocks(HH * HH), TPB, 0, stream>>>(W3, WT);
    deg_kernel<<<rowBlocks(BB * n), TPB, 0, stream>>>(ak2, dvec, BB * n, n);
    xw_gemm_kernel<<<waveBlocks(BB * n / 16), TPB, 0, stream>>>(hk2, WT, xwT, BB * n, n, npad);
    const int MT = (n + 15) / 16;
    agg_gemm_kernel<<<waveBlocks(BB * MT), TPB, 0, stream>>>(ak2, dvec, xwT, b3, hbuf, BB, n, npad);
    score_tu_kernel<<<rowBlocks(BB * n), TPB, 0, stream>>>(hbuf, pr3, po3, t, u, BB * n);
    score_final_kernel<<<rowBlocks(BB * n), TPB, 0, stream>>>(ak2, t, u, pb3, s, BB * n, n);
    topk_kernel<<<BB, TPB, 0, stream>>>(s, vals, idx, n, k);
    gather_h_kernel<<<thrBlocks(BB * k * HH), TPB, 0, stream>>>(hbuf, idx, vals, hk3, n, k);
    gather_a_kernel<<<thrBlocks(BB * k * k), TPB, 0, stream>>>(ak2, idx, ak3, n, k);
    readout_kernel<<<thrBlocks(BB * HH), TPB, 0, stream>>>(hk3, rr, k);
  }

  // ================= MLP head =================
  mlp_kernel<<<BB, TPB, 0, stream>>>(rr, l1w, l1b, l2w, l2b, l3w, l3b, out);
}
